// CondGruAttentionSeq_30855045054501
// MI455X (gfx1250) — matrix operand order repaired and checked by reading, not compiled, not run on hardware
//
#include <hip/hip_runtime.h>

typedef __attribute__((ext_vector_type(16))) __bf16 v16bf;
typedef __attribute__((ext_vector_type(8)))  __bf16 v8bf;
typedef __attribute__((ext_vector_type(8)))  float  v8f;

#define VV 50257
#define HH 512
#define EMBD 256
#define EE 128
#define TT 64
#define BB 32
#define TCC 32
#define SSQ 35            // 3 cond + 32 seq
#define KFEAT 896         // E + EMB + H
#define NCOMB 2048        // 512 attn-query + 1536 gates
static const long PREDN = (long)TT * BB * VV;

// ---------------------------------------------------------------------------
// bf16 fragment helpers (16x16x32 WMMA; k order chosen here, same for A and B)
//   A: lane = M row (lane&15), contiguous 16-K half selected by lane[4]
//   B: lane = N col (lane&15), contiguous 16-K half selected by lane[4]
// ---------------------------------------------------------------------------
__device__ inline v16bf make_a_frag(const __bf16* __restrict__ p) {
  v8bf lo = *(const v8bf*)(p);
  v8bf hi = *(const v8bf*)(p + 8);
  v16bf f;
#pragma unroll
  for (int i = 0; i < 8; ++i) { f[i] = lo[i]; f[i + 8] = hi[i]; }
  return f;
}

// ---------------------------------------------------------------------------
// generic bf16 WMMA GEMM:  C[m,n] = sum_k A[m,k]*B[n,k] + bias[n]
// grid = (ceil(N/128), M/128), block = 256 (8 waves, 2x4 wave tiling)
// Branch-free inner loop: OOB columns are CLAMPED (never stored), so the
// K-loop is pure loads + v_wmma with no exec-mask manipulation.
// nt_store=1 -> non-temporal C stores (streaming output, keep B in L2).
// ---------------------------------------------------------------------------
__global__ __launch_bounds__(256) void k_gemm_bf16(
    const __bf16* __restrict__ A, int lda,
    const __bf16* __restrict__ B, int ldb,
    const float* __restrict__ bias,
    float* __restrict__ C, int ldc, int N, int K, int nt_store)
{
  const int lane = threadIdx.x & 31;
  const int w    = threadIdx.x >> 5;
  const int mh   = w >> 2;          // 0..1  (64-row half)
  const int nq   = w & 3;           // 0..3  (32-col quarter)
  const int blockM = blockIdx.y * 128;
  const int blockN = blockIdx.x * 128;

  // hoisted lane-resolved base pointers
  const int koA = (lane & 16) ? 16 : 0;
  const int koB = (lane & 16) ? 16 : 0;
  const __bf16* aptr[4];
#pragma unroll
  for (int i = 0; i < 4; ++i) {
    int row = blockM + mh * 64 + i * 16 + (lane & 15);
    aptr[i] = A + (long)row * lda + koA;
  }
  const __bf16* bptr[2];
#pragma unroll
  for (int j = 0; j < 2; ++j) {
    int col = blockN + nq * 32 + j * 16 + (lane & 15);
    col = min(col, N - 1);                 // clamp: OOB lanes re-read valid row
    bptr[j] = B + (long)col * ldb + koB;
  }

  v8f zero = {};
  v8f acc[4][2];
#pragma unroll
  for (int i = 0; i < 4; ++i)
#pragma unroll
    for (int j = 0; j < 2; ++j) acc[i][j] = zero;

  for (int kb = 0; kb < K; kb += 32) {
    // issue B first (shared by all M-subtiles), then A in consume order
    v16bf bfr[2];
#pragma unroll
    for (int j = 0; j < 2; ++j) bfr[j] = *(const v16bf*)(bptr[j] + kb);
    v16bf afr[4];
#pragma unroll
    for (int i = 0; i < 4; ++i) afr[i] = make_a_frag(aptr[i] + kb);
    // consume in load-issue order so waits retire incrementally
#pragma unroll
    for (int i = 0; i < 4; ++i)
#pragma unroll
      for (int j = 0; j < 2; ++j)
        acc[i][j] = __builtin_amdgcn_wmma_f32_16x16x32_bf16(
            false, afr[i], false, bfr[j], (short)0, acc[i][j], false, false);
  }

  const int mo   = (lane & 16) ? 8 : 0;
  const int ncol = lane & 15;
#pragma unroll
  for (int i = 0; i < 4; ++i)
#pragma unroll
    for (int j = 0; j < 2; ++j) {
      int n = blockN + nq * 32 + j * 16 + ncol;
      if (n >= N) continue;
      float bv = bias ? bias[n] : 0.f;
      if (nt_store) {
#pragma unroll
        for (int r = 0; r < 8; ++r) {
          int m = blockM + mh * 64 + i * 16 + mo + r;
          __builtin_nontemporal_store(acc[i][j][r] + bv, &C[(long)m * ldc + n]);
        }
      } else {
#pragma unroll
        for (int r = 0; r < 8; ++r) {
          int m = blockM + mh * 64 + i * 16 + mo + r;
          C[(long)m * ldc + n] = acc[i][j][r] + bv;
        }
      }
    }
}

// ---------------------------------------------------------------------------
// small prep kernels
// ---------------------------------------------------------------------------
__global__ void k_convert(const float* __restrict__ in, __bf16* __restrict__ out, long n) {
  long i = (long)blockIdx.x * blockDim.x + threadIdx.x;
  long stride = (long)gridDim.x * blockDim.x;
  for (; i < n; i += stride) out[i] = (__bf16)in[i];
}

// Wcomb[n,k]: n<512 -> W_h = attn_W[n, 128+k]; else gru_Whh[n-512, k]
__global__ void k_wcomb(const float* __restrict__ attn_W,
                        const float* __restrict__ gru_Whh,
                        __bf16* __restrict__ out) {
  long i = (long)blockIdx.x * blockDim.x + threadIdx.x;
  if (i >= (long)NCOMB * HH) return;
  int n = (int)(i >> 9), k = (int)(i & 511);
  float v = (n < HH) ? attn_W[(long)n * (EE + HH) + EE + k]
                     : gru_Whh[(long)(n - HH) * HH + k];
  out[i] = (__bf16)v;
}

__global__ void k_cond_emb(const int* __restrict__ condition,
                           const float* __restrict__ e0,
                           const float* __restrict__ e1,
                           const float* __restrict__ e2,
                           float* __restrict__ enc_b) {
  int i = blockIdx.x * blockDim.x + threadIdx.x;   // 3*32*128
  if (i >= 3 * BB * EE) return;
  int e = i & 127, b = (i >> 7) & 31, s = i >> 12;
  const float* tab = (s == 0) ? e0 : (s == 1) ? e1 : e2;
  int idx = condition[s * BB + b];
  enc_b[((long)b * SSQ + s) * EE + e] = tab[(long)idx * EE + e];
}

__global__ __launch_bounds__(384) void k_cgru_xw(
    const int* __restrict__ cond_seq, const float* __restrict__ cse,
    const float* __restrict__ Wih, const float* __restrict__ bih,
    float* __restrict__ cxw) {
  __shared__ float emb[EE];
  int row = blockIdx.x;                       // t*32+b, 1024 rows
  int idx = cond_seq[row];
  if (threadIdx.x < EE) emb[threadIdx.x] = cse[(long)idx * EE + threadIdx.x];
  __syncthreads();
  int g = threadIdx.x;                        // 384
  const float* wr = Wih + (long)g * EE;
  float acc = bih[g];
  for (int k = 0; k < EE; ++k) acc += wr[k] * emb[k];
  cxw[(long)row * 384 + g] = acc;
}

__global__ __launch_bounds__(384) void k_cgru_scan(
    const float* __restrict__ cxw, const float* __restrict__ Whh,
    const float* __restrict__ bhh, float* __restrict__ enc_b) {
  __shared__ float h[EE];
  __shared__ float hw[384];
  int b = blockIdx.x;                         // 32 blocks, one per batch
  int g = threadIdx.x;
  if (g < EE) h[g] = 0.f;
  __syncthreads();
  for (int t = 0; t < TCC; ++t) {
    const float* wr = Whh + (long)g * EE;
    float a = bhh[g];
    for (int k = 0; k < EE; ++k) a += wr[k] * h[k];
    hw[g] = a;
    __syncthreads();
    if (g < EE) {
      const float* x = cxw + ((long)t * BB + b) * 384;
      float r  = 1.f / (1.f + expf(-(x[g] + hw[g])));
      float z  = 1.f / (1.f + expf(-(x[EE + g] + hw[EE + g])));
      float n  = tanhf(x[2 * EE + g] + r * hw[2 * EE + g]);
      float hn = (1.f - z) * n + z * h[g];
      h[g] = hn;
      enc_b[((long)b * SSQ + (3 + t)) * EE + g] = hn;
    }
    __syncthreads();
  }
}

__global__ __launch_bounds__(512) void k_enc_proj(
    const float* __restrict__ enc_b, const float* __restrict__ attn_W,
    float* __restrict__ enc_proj) {
  __shared__ float er[EE];
  long bs = blockIdx.x;                       // 1120 = b*35+s
  if (threadIdx.x < EE) er[threadIdx.x] = enc_b[bs * EE + threadIdx.x];
  __syncthreads();
  int hcol = threadIdx.x;                     // 512
  const float* wr = attn_W + (long)hcol * (EE + HH);   // first 128 cols = W_e
  float a = 0.f;
  for (int k = 0; k < EE; ++k) a += wr[k] * er[k];
  enc_proj[bs * HH + hcol] = a;
}

__global__ __launch_bounds__(256) void k_embed(
    const int* __restrict__ input_data, const float* __restrict__ embedding,
    __bf16* __restrict__ xembbf) {
  int row = blockIdx.x;                       // 2048
  int idx = input_data[row];
  int j = threadIdx.x;                        // 256
  xembbf[(long)row * EMBD + j] = (__bf16)embedding[(long)idx * EMBD + j];
}

// ---------------------------------------------------------------------------
// main attention-GRU recurrence: 2 blocks x 16 batches, WMMA per step for
// G = h x [W_h | Whh]^T  ([16,512]x[512,2048]), VALU attention/gates.
// ---------------------------------------------------------------------------
__global__ __launch_bounds__(512) void k_recurrence(
    const float* __restrict__ enc_b,     // [32][35][128]
    const float* __restrict__ enc_proj,  // [32][35][512]
    const float* __restrict__ xw,        // [2048][1536] (includes bih)
    const __bf16* __restrict__ xembbf,   // [2048][256]
    const __bf16* __restrict__ Wcomb,    // [2048][512] bf16
    const float* __restrict__ attn_b,    // [512]
    const float* __restrict__ vW,        // [512]
    const float* __restrict__ gru_bhh,   // [1536]
    __bf16* __restrict__ feat,           // [2048][896]
    float* __restrict__ out_w,           // [64][32][35]
    float* __restrict__ out_h)           // [32][512]
{
  extern __shared__ char smem[];
  float*  Gs   = (float*)smem;                 // 16*2048 f32  (131072 B)
  float*  hbuf = Gs + 16 * NCOMB;              // 16*512  f32
  float*  sc   = hbuf + 16 * HH;               // 16*36   f32
  __bf16* hbf  = (__bf16*)(sc + 16 * 36);      // 16*512  bf16

  const int tid  = threadIdx.x;
  const int lane = tid & 31;
  const int wv   = tid >> 5;                   // 16 waves
  const int bbase = blockIdx.x * 16;           // batch-group base

  for (int i = tid; i < 16 * HH; i += 512) { hbuf[i] = 0.f; hbf[i] = (__bf16)0.f; }
  __syncthreads();

  // hoisted lane pointers for phase-1 fragments
  const __bf16* aP = hbf + (lane & 15) * HH + ((lane & 16) ? 16 : 0);
  const __bf16* bP[8];
#pragma unroll
  for (int nt = 0; nt < 8; ++nt) {
    int col = wv * 128 + nt * 16 + (lane & 15);
    bP[nt] = Wcomb + (long)col * HH + ((lane & 16) ? 16 : 0);
  }

  for (int t = 0; t < TT; ++t) {
    // ---- phase 1: G = h x Wcomb^T (WMMA) ----
    {
      v8f zero = {};
      v8f acc[8];
#pragma unroll
      for (int nt = 0; nt < 8; ++nt) acc[nt] = zero;
      for (int ks = 0; ks < 16; ++ks) {
        const int kb = ks * 32;
        v16bf afr = make_a_frag(aP + kb);
#pragma unroll
        for (int nt = 0; nt < 8; ++nt) {
          v16bf bfr = *(const v16bf*)(bP[nt] + kb);
          acc[nt] = __builtin_amdgcn_wmma_f32_16x16x32_bf16(
              false, afr, false, bfr, (short)0, acc[nt], false, false);
        }
      }
      const int mo = (lane & 16) ? 8 : 0;
      const int n0 = wv * 128;
#pragma unroll
      for (int nt = 0; nt < 8; ++nt) {
        int col = n0 + nt * 16 + (lane & 15);
#pragma unroll
        for (int r = 0; r < 8; ++r) Gs[(mo + r) * NCOMB + col] = acc[nt][r];
      }
    }
    __syncthreads();

    // ---- phase 2a: attention scores (wave wv owns local batch wv) ----
    {
      int b = wv;
      for (int s = 0; s < SSQ; ++s) {
        const float* ep = enc_proj + (((long)(bbase + b)) * SSQ + s) * HH;
        float p = 0.f;
        for (int hc = lane; hc < HH; hc += 32) {
          float q = Gs[b * NCOMB + hc] + attn_b[hc];
          p += tanhf(ep[hc] + q) * vW[hc];
        }
#pragma unroll
        for (int off = 16; off; off >>= 1) p += __shfl_xor(p, off, 32);
        if (lane == 0) sc[b * 36 + s] = p;
      }
    }
    __syncthreads();

    // ---- phase 2b: softmax over s + weights output ----
    if (tid < 16) {
      int b = tid;
      float m = -3.4e38f;
      for (int s = 0; s < SSQ; ++s) m = fmaxf(m, sc[b * 36 + s]);
      float ssum = 0.f;
      for (int s = 0; s < SSQ; ++s) ssum += expf(sc[b * 36 + s] - m);
      float inv = 1.f / ssum;
      for (int s = 0; s < SSQ; ++s) {
        float wvv = expf(sc[b * 36 + s] - m) * inv;
        sc[b * 36 + s] = wvv;
        out_w[((long)t * BB + bbase + b) * SSQ + s] = wvv;
      }
    }
    __syncthreads();

    // ---- phase 2c: context + build bf16 feature row [c | xe | h_prev] ----
    const long row_base = (long)t * BB + bbase;
    for (int i = 0; i < 4; ++i) {
      int idx = tid + 512 * i;                 // 2048 = 16*128
      int b = idx >> 7, e = idx & 127;
      const float* eb = enc_b + ((long)(bbase + b)) * SSQ * EE + e;
      float cv = 0.f;
      for (int s = 0; s < SSQ; ++s) cv += sc[b * 36 + s] * eb[s * EE];
      feat[(row_base + b) * KFEAT + e] = (__bf16)cv;
    }
    for (int i = 0; i < 8; ++i) {
      int idx = tid + 512 * i;                 // 4096 = 16*256
      int b = idx >> 8, j = idx & 255;
      feat[(row_base + b) * KFEAT + EE + j] = xembbf[(row_base + b) * EMBD + j];
    }
    for (int i = 0; i < 16; ++i) {
      int idx = tid + 512 * i;                 // 8192 = 16*512
      int b = idx >> 9, g = idx & 511;
      feat[(row_base + b) * KFEAT + EE + EMBD + g] = (__bf16)hbuf[b * HH + g];
    }

    // ---- phase 2d: GRU gates -> hn in registers ----
    float hn[16];
#pragma unroll
    for (int i = 0; i < 16; ++i) {
      int idx = tid + 512 * i;
      int b = idx >> 9, g = idx & 511;
      const float* xwr = xw + (row_base + b) * 1536;
      float hwr = Gs[b * NCOMB + HH + g]        + gru_bhh[g];
      float hwz = Gs[b * NCOMB + 2 * HH + g]    + gru_bhh[HH + g];
      float hwn = Gs[b * NCOMB + 3 * HH + g]    + gru_bhh[2 * HH + g];
      float r = 1.f / (1.f + expf(-(xwr[g] + hwr)));
      float z = 1.f / (1.f + expf(-(xwr[HH + g] + hwz)));
      float n = tanhf(xwr[2 * HH + g] + r * hwn);
      hn[i] = (1.f - z) * n + z * hbuf[b * HH + g];
    }
    __syncthreads();
#pragma unroll
    for (int i = 0; i < 16; ++i) {
      int idx = tid + 512 * i;
      int b = idx >> 9, g = idx & 511;
      hbuf[b * HH + g] = hn[i];
      hbf[b * HH + g]  = (__bf16)hn[i];
    }
    __syncthreads();
  }

  for (int i = 0; i < 16; ++i) {
    int idx = tid + 512 * i;
    int b = idx >> 9, g = idx & 511;
    out_h[(long)(bbase + b) * HH + g] = hbuf[b * HH + g];
  }
}

// ---------------------------------------------------------------------------
// row-wise log-softmax over V (online max/sum, then write-back from L2;
// final writes are non-temporal: last touch of the output stream)
// ---------------------------------------------------------------------------
__global__ __launch_bounds__(256) void k_logsoftmax(float* __restrict__ pred) {
  __shared__ float mb[8], sb[8];
  __shared__ float lse_s;
  long row = blockIdx.x;
  float* p = pred + row * (long)VV;
  const int lane = threadIdx.x & 31;
  const int wv   = threadIdx.x >> 5;          // 8 waves
  float m = -3.4e38f, s = 0.f;
  for (int v = threadIdx.x; v < VV; v += 256) {
    float x = p[v];
    if (x > m) { s = s * expf(m - x) + 1.f; m = x; }
    else       { s += expf(x - m); }
  }
  // wave-level combine of (m, s)
#pragma unroll
  for (int off = 16; off; off >>= 1) {
    float mo = __shfl_xor(m, off, 32);
    float so = __shfl_xor(s, off, 32);
    float M  = fmaxf(m, mo);
    s = s * expf(m - M) + so * expf(mo - M);
    m = M;
  }
  if (lane == 0) { mb[wv] = m; sb[wv] = s; }
  __syncthreads();
  if (threadIdx.x == 0) {
    float M = -3.4e38f;
#pragma unroll
    for (int i = 0; i < 8; ++i) M = fmaxf(M, mb[i]);
    float S = 0.f;
#pragma unroll
    for (int i = 0; i < 8; ++i) S += sb[i] * expf(mb[i] - M);
    lse_s = M + logf(S);
  }
  __syncthreads();
  float lse = lse_s;
  for (int v = threadIdx.x; v < VV; v += 256)
    __builtin_nontemporal_store(p[v] - lse, &p[v]);
}

// ---------------------------------------------------------------------------
extern "C" void kernel_launch(void* const* d_in, const int* in_sizes, int n_in,
                              void* d_out, int out_size, void* d_ws, size_t ws_size,
                              hipStream_t stream) {
  (void)in_sizes; (void)n_in; (void)out_size; (void)ws_size;
  const int*   input_data = (const int*)d_in[0];
  const int*   condition  = (const int*)d_in[1];
  const int*   cond_seq   = (const int*)d_in[2];
  const float* emb0       = (const float*)d_in[3];
  const float* emb1       = (const float*)d_in[4];
  const float* emb2       = (const float*)d_in[5];
  const float* cse        = (const float*)d_in[6];
  const float* embedding  = (const float*)d_in[7];
  const float* attn_W     = (const float*)d_in[8];
  const float* attn_b     = (const float*)d_in[9];
  const float* v_W        = (const float*)d_in[10];
  const float* gru_Wih    = (const float*)d_in[11];
  const float* gru_Whh    = (const float*)d_in[12];
  const float* gru_bih    = (const float*)d_in[13];
  const float* gru_bhh    = (const float*)d_in[14];
  const float* cgru_Wih   = (const float*)d_in[15];
  const float* cgru_Whh   = (const float*)d_in[16];
  const float* cgru_bih   = (const float*)d_in[17];
  const float* cgru_bhh   = (const float*)d_in[18];
  const float* dense_W    = (const float*)d_in[19];
  const float* dense_b    = (const float*)d_in[20];

  float* out   = (float*)d_out;
  float* pred  = out;                         // [64,32,50257]
  float* out_h = out + PREDN;                 // [1,32,512]
  float* out_w = out_h + (long)BB * HH;       // [64,32,35]

  // carve workspace (256B aligned)
  char* ws = (char*)d_ws;
  auto carve = [&](size_t bytes) -> char* {
    char* p = ws; ws += (bytes + 255) & ~(size_t)255; return p;
  };
  float*  enc_b    = (float*) carve((size_t)BB * SSQ * EE * 4);
  float*  enc_proj = (float*) carve((size_t)BB * SSQ * HH * 4);
  float*  cxw      = (float*) carve((size_t)TCC * BB * 384 * 4);
  float*  xw       = (float*) carve((size_t)TT * BB * 1536 * 4);
  __bf16* xembbf   = (__bf16*)carve((size_t)TT * BB * EMBD * 2);
  __bf16* wihbf    = (__bf16*)carve((size_t)1536 * EMBD * 2);
  __bf16* wcomb    = (__bf16*)carve((size_t)NCOMB * HH * 2);
  __bf16* featbf   = (__bf16*)carve((size_t)TT * BB * KFEAT * 2);
  __bf16* wdense   = (__bf16*)carve((size_t)VV * KFEAT * 2);

  // precision conversions / weight packing
  k_convert<<<4096, 256, 0, stream>>>(dense_W, wdense, (long)VV * KFEAT);
  k_convert<<<256,  256, 0, stream>>>(gru_Wih, wihbf, (long)1536 * EMBD);
  k_wcomb<<<(NCOMB * HH + 255) / 256, 256, 0, stream>>>(attn_W, gru_Whh, wcomb);

  // encoder: condition embeddings + conditional GRU
  k_cond_emb<<<48, 256, 0, stream>>>(condition, emb0, emb1, emb2, enc_b);
  k_cgru_xw<<<TCC * BB, 384, 0, stream>>>(cond_seq, cse, cgru_Wih, cgru_bih, cxw);
  k_cgru_scan<<<BB, 384, 0, stream>>>(cxw, cgru_Whh, cgru_bhh, enc_b);
  k_enc_proj<<<BB * SSQ, 512, 0, stream>>>(enc_b, attn_W, enc_proj);

  // decoder inputs: token embedding gather + input projection (WMMA GEMM)
  k_embed<<<TT * BB, 256, 0, stream>>>(input_data, embedding, xembbf);
  k_gemm_bf16<<<dim3(1536 / 128, (TT * BB) / 128), 256, 0, stream>>>(
      xembbf, EMBD, wihbf, EMBD, gru_bih, xw, 1536, 1536, EMBD, /*nt=*/0);

  // sequential attention-GRU scan (2 batch-groups, WMMA per step)
  size_t smem = (size_t)(16 * NCOMB + 16 * HH + 16 * 36) * 4 + (size_t)16 * HH * 2;
  k_recurrence<<<2, 512, smem, stream>>>(enc_b, enc_proj, xw, xembbf, wcomb,
                                         attn_b, v_W, gru_bhh, featbf, out_w, out_h);

  // dominant GEMM: [2048, 896] x dense_W^T -> logits
  // (bf16 WMMA; W stays L2-resident, logits streamed out non-temporally)
  k_gemm_bf16<<<dim3((VV + 127) / 128, (TT * BB) / 128), 256, 0, stream>>>(
      featbf, KFEAT, wdense, KFEAT, dense_b, pred, VV, VV, KFEAT, /*nt=*/1);

  // fused log-softmax
  k_logsoftmax<<<TT * BB, 256, 0, stream>>>(pred);
}
